// CubeSolver_15178414424459
// MI455X (gfx1250) — matrix operand order repaired and checked by reading, not compiled, not run on hardware
//
#include <hip/hip_runtime.h>
#include <cstdint>
#include <cstddef>

// ---------------------------------------------------------------------------
// Types for CDNA5 WMMA (wave32): v_wmma_f32_16x16x32_f16
// ---------------------------------------------------------------------------
typedef _Float16 v16h __attribute__((ext_vector_type(16)));
typedef _Float16 h8   __attribute__((ext_vector_type(8)));
typedef float    v8f  __attribute__((ext_vector_type(8)));

union FragAB { v16h v; h8 h[2]; };

// ---------------------------------------------------------------------------
// Elementwise helpers
// ---------------------------------------------------------------------------
__global__ void k_cvt_f16(const float* __restrict__ s, _Float16* __restrict__ d, size_t n) {
    size_t i = (size_t)blockIdx.x * 256 + threadIdx.x;
    if (i < n) d[i] = (_Float16)s[i];
}

__global__ void k_zero_u32(uint32_t* __restrict__ p, size_t n) {
    size_t i = (size_t)blockIdx.x * 256 + threadIdx.x;
    if (i < n) p[i] = 0u;
}

// xf += delta ; write f16 mirror
__global__ void k_add_resid(float* __restrict__ xf, const float* __restrict__ delta,
                            _Float16* __restrict__ x16, size_t n) {
    size_t i = (size_t)blockIdx.x * 256 + threadIdx.x;
    if (i >= n) return;
    float v = xf[i] + delta[i];
    xf[i] = v;
    x16[i] = (_Float16)v;
}

// ---------------------------------------------------------------------------
// GEMM: C[M,N] = act(A[M,K] @ W[N,K]^T + bias)  (A,W f16; C f32 and/or f16)
//
// Block: 256 threads (8 waves). Tile BM=128, BN=128, K-step 32.
// Waves: wm = w&1 (2 along M, 64 rows each), wn = w>>1 (4 along N, 32 cols
// each). Each wave: 4x2 grid of v_wmma_f32_16x16x32_f16 -> 8 WMMA / K-step.
//
// Tiles are streamed global->LDS with gfx1250 async-to-LDS copies
// (global_load_async_to_lds_b128, ASYNCcnt) into double-buffered LDS, so
// tile k+1 transfers overlap the WMMAs of tile k.
//
// Out-of-range B rows (N=21 head, N=64 dbl) are clamped to row N-1: their
// products only reach output columns n>=N, which are masked at store.
//
// act: 0 = none, 1 = exact GELU
// ---------------------------------------------------------------------------
__global__ __launch_bounds__(256) void k_gemm_wmma(
    const _Float16* __restrict__ A, const _Float16* __restrict__ W,
    const float* __restrict__ bias, float* __restrict__ outF,
    _Float16* __restrict__ outH, int M, int N, int K, int act)
{
    // 40-halve row stride: 80B rows keep every 16B chunk 16B-aligned
    __shared__ __align__(16) _Float16 sA[2][128][40];
    __shared__ __align__(16) _Float16 sB[2][128][40];
    constexpr uint32_t BUFSTRIDE = 128u * 40u * 2u;   // 10240 bytes

    const int tid  = threadIdx.x;
    const int lane = tid & 31;
    const int w    = tid >> 5;
    const int wm   = w & 1;           // 0..1  (64 rows each)
    const int wn   = w >> 1;          // 0..3  (32 cols each)
    const int bm   = blockIdx.y * 128;
    const int bn   = blockIdx.x * 128;

    const int halfsel = lane >> 4;    // 0: lanes 0-15, 1: lanes 16-31
    const int lmod    = lane & 15;

    // ---- per-thread staging addresses (4x 16B chunks per K-step) ----
    const int rowL = tid >> 2;            // 0..63
    const int kc   = (tid & 3) * 8;       // halve offset {0,8,16,24}

    uint32_t ldsA0 = (uint32_t)(uintptr_t)&sA[0][rowL][kc];
    uint32_t ldsA1 = (uint32_t)(uintptr_t)&sA[0][rowL + 64][kc];
    uint32_t ldsB0 = (uint32_t)(uintptr_t)&sB[0][rowL][kc];
    uint32_t ldsB1 = (uint32_t)(uintptr_t)&sB[0][rowL + 64][kc];

    int nb0 = bn + rowL;        if (nb0 > N - 1) nb0 = N - 1;
    int nb1 = bn + rowL + 64;   if (nb1 > N - 1) nb1 = N - 1;

    uint64_t gA0 = (uint64_t)(uintptr_t)(A + (size_t)(bm + rowL) * K + kc);
    uint64_t gA1 = (uint64_t)(uintptr_t)(A + (size_t)(bm + rowL + 64) * K + kc);
    uint64_t gB0 = (uint64_t)(uintptr_t)(W + (size_t)nb0 * K + kc);
    uint64_t gB1 = (uint64_t)(uintptr_t)(W + (size_t)nb1 * K + kc);

    auto issue_tile = [&](int buf) {
        const uint32_t bo = (uint32_t)buf * BUFSTRIDE;
        asm volatile("global_load_async_to_lds_b128 %0, %1, off"
                     :: "v"(ldsA0 + bo), "v"(gA0) : "memory");
        asm volatile("global_load_async_to_lds_b128 %0, %1, off"
                     :: "v"(ldsA1 + bo), "v"(gA1) : "memory");
        asm volatile("global_load_async_to_lds_b128 %0, %1, off"
                     :: "v"(ldsB0 + bo), "v"(gB0) : "memory");
        asm volatile("global_load_async_to_lds_b128 %0, %1, off"
                     :: "v"(ldsB1 + bo), "v"(gB1) : "memory");
        gA0 += 64; gA1 += 64; gB0 += 64; gB1 += 64;   // next K-step (32 halves)
    };

    v8f acc[4][2];
    #pragma unroll
    for (int a = 0; a < 4; a++)
        #pragma unroll
        for (int b = 0; b < 2; b++)
            #pragma unroll
            for (int q = 0; q < 8; q++) acc[a][b][q] = 0.0f;

    issue_tile(0);
    int buf = 0;

    for (int k0 = 0; k0 < K; k0 += 32) {
        asm volatile("s_wait_asynccnt 0x0" ::: "memory");
        __syncthreads();                       // tile `buf` visible to all waves
        if (k0 + 32 < K) issue_tile(buf ^ 1);  // overlap next transfer with WMMAs

        const _Float16 (*__restrict__ cA)[40] = sA[buf];
        const _Float16 (*__restrict__ cB)[40] = sB[buf];

        // B fragments (32x16 f16): lanes 0-15 hold K{0..7,16..23} (col n = lane),
        // lanes 16-31 hold K{8..15,24..31}
        FragAB bf[2];
        {
            const int kb = halfsel * 8;
            #pragma unroll
            for (int ni = 0; ni < 2; ni++) {
                const int r = wn * 32 + ni * 16 + lmod;
                bf[ni].h[0] = *(const h8*)&cB[r][kb];
                bf[ni].h[1] = *(const h8*)&cB[r][kb + 16];
            }
        }
        // A fragments (16x32 f16, ISA 7.12.2): lane<16 holds K{0..7,16..23},
        // lane>=16 holds K{8..15,24..31}
        {
            const int ka = halfsel * 8;
            #pragma unroll
            for (int mi = 0; mi < 4; mi++) {
                const int r = wm * 64 + mi * 16 + lmod;
                FragAB af;
                af.h[0] = *(const h8*)&cA[r][ka];
                af.h[1] = *(const h8*)&cA[r][ka + 16];
                acc[mi][0] = __builtin_amdgcn_wmma_f32_16x16x32_f16(
                    false, af.v, false, bf[0].v, (short)0, acc[mi][0], false, false);
                acc[mi][1] = __builtin_amdgcn_wmma_f32_16x16x32_f16(
                    false, af.v, false, bf[1].v, (short)0, acc[mi][1], false, false);
            }
        }
        buf ^= 1;
    }

    // --- epilogue: C layout m = r + 8*(lane>>4), n = lane&15 ---
    #pragma unroll
    for (int mi = 0; mi < 4; mi++) {
        #pragma unroll
        for (int ni = 0; ni < 2; ni++) {
            const int mbase = bm + wm * 64 + mi * 16 + halfsel * 8;
            const int n     = bn + wn * 32 + ni * 16 + lmod;
            if (n < N) {
                const float bv = bias ? bias[n] : 0.0f;
                #pragma unroll
                for (int r = 0; r < 8; r++) {
                    float v = acc[mi][ni][r] + bv;
                    if (act == 1) v = 0.5f * v * (1.0f + erff(v * 0.70710678118654752f));
                    const size_t o = (size_t)(mbase + r) * N + n;
                    if (outF) outF[o] = v;
                    if (outH) outH[o] = (_Float16)v;
                }
            }
        }
    }
}

// ---------------------------------------------------------------------------
// LayerNorm over D=512 (one block of 256 threads per row), optional residual,
// optional f32/f16 outputs.
// ---------------------------------------------------------------------------
__global__ __launch_bounds__(256) void k_ln512(
    const float* __restrict__ in, const float* __restrict__ res,
    const float* __restrict__ g, const float* __restrict__ b,
    float* __restrict__ o32, _Float16* __restrict__ o16)
{
    __shared__ float red[256];
    __shared__ float stats[2];
    const int row = blockIdx.x, t = threadIdx.x;
    const size_t base = (size_t)row * 512;

    float a0 = in[base + t], a1 = in[base + t + 256];
    if (res) { a0 += res[base + t]; a1 += res[base + t + 256]; }

    red[t] = a0 + a1;
    __syncthreads();
    for (int s = 128; s > 0; s >>= 1) { if (t < s) red[t] += red[t + s]; __syncthreads(); }
    if (t == 0) stats[0] = red[0] * (1.0f / 512.0f);
    __syncthreads();
    const float mu = stats[0];
    const float d0 = a0 - mu, d1 = a1 - mu;
    red[t] = d0 * d0 + d1 * d1;
    __syncthreads();
    for (int s = 128; s > 0; s >>= 1) { if (t < s) red[t] += red[t + s]; __syncthreads(); }
    if (t == 0) stats[1] = rsqrtf(red[0] * (1.0f / 512.0f) + 1e-5f);
    __syncthreads();
    const float rs = stats[1];

    const float r0 = d0 * rs * g[t] + b[t];
    const float r1 = d1 * rs * g[t + 256] + b[t + 256];
    if (o32) { o32[base + t] = r0; o32[base + t + 256] = r1; }
    if (o16) { o16[base + t] = (_Float16)r0; o16[base + t + 256] = (_Float16)r1; }
}

// ---------------------------------------------------------------------------
// GRU gate fusion: z_new from gi[M,1536], gh[M,1536], z[M,512]
// ---------------------------------------------------------------------------
__global__ void k_gru_gates(const float* __restrict__ gi, const float* __restrict__ gh,
                            float* __restrict__ z32, _Float16* __restrict__ z16, int total)
{
    const int i = blockIdx.x * 256 + threadIdx.x;
    if (i >= total) return;
    const int m = i >> 9, d = i & 511;
    const size_t b3 = (size_t)m * 1536;
    const float ir = gi[b3 + d],       hr = gh[b3 + d];
    const float iz = gi[b3 + 512 + d], hz = gh[b3 + 512 + d];
    const float in_ = gi[b3 + 1024 + d], hn = gh[b3 + 1024 + d];
    const float r  = 1.0f / (1.0f + expf(-(ir + hr)));
    const float u  = 1.0f / (1.0f + expf(-(iz + hz)));
    const float nn = tanhf(in_ + r * hn);
    const float zn = (1.0f - u) * nn + u * z32[i];
    z32[i] = zn;
    z16[i] = (_Float16)zn;
}

// ---------------------------------------------------------------------------
// Mamba depthwise causal conv (DC=4) + conv_b + SiLU; xi = xz[..., :1024]
// ---------------------------------------------------------------------------
__global__ void k_conv_silu(const float* __restrict__ xz, const float* __restrict__ cw,
                            const float* __restrict__ cb, float* __restrict__ xc32,
                            _Float16* __restrict__ xc16, int total)
{
    const int i = blockIdx.x * 256 + threadIdx.x;
    if (i >= total) return;
    const int di = i & 1023;
    const int m  = i >> 10;
    const int l  = m & 2047;
    float acc = cb[di];
    #pragma unroll
    for (int j = 0; j < 4; j++) {
        const int l2 = l + j - 3;
        if (l2 >= 0) acc += xz[(size_t)(m + j - 3) * 2048 + di] * cw[di * 4 + j];
    }
    const float s = acc / (1.0f + expf(-acc));   // SiLU
    xc32[i] = s;
    xc16[i] = (_Float16)s;
}

// ---------------------------------------------------------------------------
// dt projection: softplus( dbl[:, :32] @ dt_w[di,:32]^T + dt_b[di] )
// ---------------------------------------------------------------------------
__global__ void k_dtproj(const float* __restrict__ dbl, const float* __restrict__ dtw,
                         const float* __restrict__ dtb, float* __restrict__ dt32, int total)
{
    const int i = blockIdx.x * 256 + threadIdx.x;
    if (i >= total) return;
    const int di = i & 1023;
    const int m  = i >> 10;
    const float* __restrict__ dr = dbl + (size_t)m * 64;
    const float* __restrict__ wr = dtw + di * 32;
    float a = dtb[di];
    #pragma unroll
    for (int r = 0; r < 32; r++) a += dr[r] * wr[r];
    dt32[i] = (a > 20.0f) ? a : log1pf(expf(a));
}

// ---------------------------------------------------------------------------
// Selective scan: per (b,di) thread, sequential over L=2048; B_t/C_t (32
// floats per step, shared across all di) staged through LDS.
// grid = B * (DI/256) = 16 blocks of 256 threads
// ---------------------------------------------------------------------------
__global__ __launch_bounds__(256) void k_scan(
    const float* __restrict__ dbl, const float* __restrict__ dt,
    const float* __restrict__ xc, const float* __restrict__ Alog,
    const float* __restrict__ Dp, float* __restrict__ yraw)
{
    const int b  = blockIdx.x >> 2;
    const int di = ((blockIdx.x & 3) << 8) + threadIdx.x;

    float Ar[16];
    #pragma unroll
    for (int s = 0; s < 16; s++) Ar[s] = -expf(Alog[di * 16 + s]);
    const float Dv = Dp[di];
    float h[16];
    #pragma unroll
    for (int s = 0; s < 16; s++) h[s] = 0.0f;

    __shared__ float sBC[32];
    for (int l = 0; l < 2048; l++) {
        const size_t m = (size_t)b * 2048 + l;
        __syncthreads();
        if (threadIdx.x < 32) sBC[threadIdx.x] = dbl[m * 64 + 32 + threadIdx.x];
        __syncthreads();
        const float dv = dt[m * 1024 + di];
        const float xv = xc[m * 1024 + di];
        const float dx = dv * xv;
        float y = 0.0f;
        #pragma unroll
        for (int s = 0; s < 16; s++) {
            h[s] = expf(dv * Ar[s]) * h[s] + dx * sBC[s];
            y += h[s] * sBC[16 + s];
        }
        yraw[m * 1024 + di] = y + xv * Dv;
    }
}

// ---------------------------------------------------------------------------
// y = yraw * silu(zgate) where zgate = xz[..., 1024+di]; f16 output for GEMM
// ---------------------------------------------------------------------------
__global__ void k_ygate(const float* __restrict__ yraw, const float* __restrict__ xz,
                        _Float16* __restrict__ y16, int total)
{
    const int i = blockIdx.x * 256 + threadIdx.x;
    if (i >= total) return;
    const int di = i & 1023;
    const int m  = i >> 10;
    const float zg = xz[(size_t)m * 2048 + 1024 + di];
    y16[i] = (_Float16)(yraw[i] * (zg / (1.0f + expf(-zg))));
}

// ---------------------------------------------------------------------------
// Host orchestration
// ---------------------------------------------------------------------------
extern "C" void kernel_launch(void* const* d_in, const int* in_sizes, int n_in,
                              void* d_out, int out_size, void* d_ws, size_t ws_size,
                              hipStream_t stream)
{
    (void)in_sizes; (void)n_in; (void)out_size; (void)ws_size;
    constexpr int Mtok = 8192;   // B*L
    constexpr int D = 512, NL = 8, DI = 1024, IN = 480, V = 21, LOOPS = 4;

    const float* x       = (const float*)d_in[0];
    const float* enc_w1  = (const float*)d_in[1];
    const float* enc_b1  = (const float*)d_in[2];
    const float* enc_w2  = (const float*)d_in[3];
    const float* enc_b2  = (const float*)d_in[4];
    const float* enc_g   = (const float*)d_in[5];
    const float* enc_beta= (const float*)d_in[6];
    const float* gru_wih = (const float*)d_in[7];
    const float* gru_whh = (const float*)d_in[8];
    const float* gru_bih = (const float*)d_in[9];
    const float* gru_bhh = (const float*)d_in[10];
    const float* prj_w1  = (const float*)d_in[11];
    const float* prj_b1  = (const float*)d_in[12];
    const float* prj_w2  = (const float*)d_in[13];
    const float* prj_b2  = (const float*)d_in[14];
    const float* trm_g   = (const float*)d_in[15];
    const float* trm_beta= (const float*)d_in[16];
    const float* in_w    = (const float*)d_in[17];
    const float* conv_w  = (const float*)d_in[18];
    const float* conv_b  = (const float*)d_in[19];
    const float* xproj_w = (const float*)d_in[20];
    const float* dt_w    = (const float*)d_in[21];
    const float* dt_b    = (const float*)d_in[22];
    const float* A_log   = (const float*)d_in[23];
    const float* Dp      = (const float*)d_in[24];
    const float* out_w   = (const float*)d_in[25];
    const float* ln_g    = (const float*)d_in[26];
    const float* ln_b    = (const float*)d_in[27];
    const float* head_w  = (const float*)d_in[28];
    const float* head_b  = (const float*)d_in[29];

    char* base = (char*)d_ws;
    size_t off = 0;
    auto alloc = [&](size_t bytes) -> void* {
        size_t o = (off + 255) & ~(size_t)255;
        off = o + bytes;
        return (void*)(base + o);
    };

    // ---- persistent arena: f16 weights + h (f32/f16) ----
    _Float16* w_enc1 = (_Float16*)alloc((size_t)D * IN * 2);
    _Float16* w_enc2 = (_Float16*)alloc((size_t)D * D * 2);
    _Float16* w_wih  = (_Float16*)alloc((size_t)3 * D * D * 2);
    _Float16* w_whh  = (_Float16*)alloc((size_t)3 * D * D * 2);
    _Float16* w_prj1 = (_Float16*)alloc((size_t)D * D * 2);
    _Float16* w_prj2 = (_Float16*)alloc((size_t)D * D * 2);
    _Float16* w_in   = (_Float16*)alloc((size_t)NL * 2 * DI * D * 2);
    _Float16* w_xprj = (_Float16*)alloc((size_t)NL * 64 * DI * 2);
    _Float16* w_out  = (_Float16*)alloc((size_t)NL * D * DI * 2);
    _Float16* w_head = (_Float16*)alloc((size_t)V * D * 2);
    float*    h32    = (float*)   alloc((size_t)Mtok * D * 4);
    _Float16* h16    = (_Float16*)alloc((size_t)Mtok * D * 2);

    const size_t mark = off;

    // ---- overlay arena A: encoder + GRU refinement ----
    _Float16* x16   = (_Float16*)alloc((size_t)Mtok * IN * 2);
    _Float16* h1_16 = (_Float16*)alloc((size_t)Mtok * D * 2);
    float*    h2_32 = (float*)   alloc((size_t)Mtok * D * 4);
    float*    xf32  = (float*)   alloc((size_t)Mtok * D * 4);
    _Float16* xf16  = (_Float16*)alloc((size_t)Mtok * D * 2);
    float*    z32   = (float*)   alloc((size_t)Mtok * D * 4);
    _Float16* z16   = (_Float16*)alloc((size_t)Mtok * D * 2);
    float*    gi32  = (float*)   alloc((size_t)Mtok * 3 * D * 4);
    float*    gh32  = (float*)   alloc((size_t)Mtok * 3 * D * 4);
    _Float16* t1_16 = (_Float16*)alloc((size_t)Mtok * D * 2);
    float*    dlt32 = (float*)   alloc((size_t)Mtok * D * 4);

    // ---- overlay arena B: mamba (reuses arena A space) ----
    off = mark;
    float*    xz    = (float*)   alloc((size_t)Mtok * 2 * DI * 4);
    float*    xc32  = (float*)   alloc((size_t)Mtok * DI * 4);
    _Float16* xc16  = (_Float16*)alloc((size_t)Mtok * DI * 2);
    float*    dbl32 = (float*)   alloc((size_t)Mtok * 64 * 4);
    float*    dt32  = (float*)   alloc((size_t)Mtok * DI * 4);
    float*    yraw  = (float*)   alloc((size_t)Mtok * DI * 4);
    _Float16* y16   = (_Float16*)alloc((size_t)Mtok * DI * 2);
    float*    mo32  = (float*)   alloc((size_t)Mtok * D * 4);

    auto cvt = [&](const float* s, _Float16* d, size_t n) {
        k_cvt_f16<<<(unsigned)((n + 255) / 256), 256, 0, stream>>>(s, d, n);
    };
    auto gemm = [&](const _Float16* A, const _Float16* W, const float* bias,
                    float* oF, _Float16* oH, int M, int N, int K, int act) {
        dim3 g((N + 127) / 128, M / 128);
        k_gemm_wmma<<<g, 256, 0, stream>>>(A, W, bias, oF, oH, M, N, K, act);
    };
    auto ln = [&](const float* in, const float* res, const float* g, const float* b,
                  float* o32, _Float16* o16) {
        k_ln512<<<Mtok, 256, 0, stream>>>(in, res, g, b, o32, o16);
    };

    // ---- weight / input f16 conversion ----
    cvt(enc_w1, w_enc1, (size_t)D * IN);
    cvt(enc_w2, w_enc2, (size_t)D * D);
    cvt(gru_wih, w_wih, (size_t)3 * D * D);
    cvt(gru_whh, w_whh, (size_t)3 * D * D);
    cvt(prj_w1, w_prj1, (size_t)D * D);
    cvt(prj_w2, w_prj2, (size_t)D * D);
    cvt(in_w,   w_in,   (size_t)NL * 2 * DI * D);
    cvt(xproj_w,w_xprj, (size_t)NL * 64 * DI);
    cvt(out_w,  w_out,  (size_t)NL * D * DI);
    cvt(head_w, w_head, (size_t)V * D);
    cvt(x, x16, (size_t)Mtok * IN);

    // ---- encoder ----
    gemm(x16, w_enc1, enc_b1, nullptr, h1_16, Mtok, D, IN, /*gelu*/1);
    gemm(h1_16, w_enc2, enc_b2, h2_32, nullptr, Mtok, D, D, 0);
    ln(h2_32, nullptr, enc_g, enc_beta, xf32, xf16);

    // ---- GRU refinement loop ----
    {
        const size_t zf = (size_t)Mtok * D;
        k_zero_u32<<<(unsigned)((zf + 255) / 256), 256, 0, stream>>>((uint32_t*)z32, zf);
        k_zero_u32<<<(unsigned)((zf / 2 + 255) / 256), 256, 0, stream>>>((uint32_t*)z16, zf / 2);
    }
    for (int it = 0; it < LOOPS; it++) {
        gemm(xf16, w_wih, gru_bih, gi32, nullptr, Mtok, 3 * D, D, 0);
        gemm(z16,  w_whh, gru_bhh, gh32, nullptr, Mtok, 3 * D, D, 0);
        k_gru_gates<<<(Mtok * D + 255) / 256, 256, 0, stream>>>(gi32, gh32, z32, z16, Mtok * D);
        gemm(z16, w_prj1, prj_b1, nullptr, t1_16, Mtok, D, D, /*gelu*/1);
        gemm(t1_16, w_prj2, prj_b2, dlt32, nullptr, Mtok, D, D, 0);
        k_add_resid<<<(Mtok * D + 255) / 256, 256, 0, stream>>>(xf32, dlt32, xf16, (size_t)Mtok * D);
    }
    ln(xf32, nullptr, trm_g, trm_beta, h32, h16);

    // ---- mamba layers ----
    for (int i = 0; i < NL; i++) {
        gemm(h16, w_in + (size_t)i * 2 * DI * D, nullptr, xz, nullptr, Mtok, 2 * DI, D, 0);
        k_conv_silu<<<(Mtok * DI + 255) / 256, 256, 0, stream>>>(
            xz, conv_w + (size_t)i * DI * 4, conv_b + (size_t)i * DI, xc32, xc16, Mtok * DI);
        gemm(xc16, w_xprj + (size_t)i * 64 * DI, nullptr, dbl32, nullptr, Mtok, 64, DI, 0);
        k_dtproj<<<(Mtok * DI + 255) / 256, 256, 0, stream>>>(
            dbl32, dt_w + (size_t)i * DI * 32, dt_b + (size_t)i * DI, dt32, Mtok * DI);
        k_scan<<<16, 256, 0, stream>>>(
            dbl32, dt32, xc32, A_log + (size_t)i * DI * 16, Dp + (size_t)i * DI, yraw);
        k_ygate<<<(Mtok * DI + 255) / 256, 256, 0, stream>>>(yraw, xz, y16, Mtok * DI);
        gemm(y16, w_out + (size_t)i * D * DI, nullptr, mo32, nullptr, Mtok, D, DI, 0);
        ln(mo32, h32, ln_g + (size_t)i * D, ln_b + (size_t)i * D, h32, h16);
    }

    // ---- head ----
    gemm(h16, w_head, head_b, (float*)d_out, nullptr, Mtok, V, D, 0);
}
